// LocationAwareAttention_33784212750984
// MI455X (gfx1250) — matrix operand order repaired and checked by reading, not compiled, not run on hardware
//
#include <hip/hip_runtime.h>

#define Bv 32
#define Tv 2048
#define Dv 1024
#define Av 1024
#define TM 32
#define LDSS 1032   // bf16 elements per LDS tile row (1024 + 8 pad -> conflict-free b128)
#define ENERGY_LDS (TM * LDSS * 2 + (TM + 2) * 4 + TM * 4)  // 66312 bytes

typedef __attribute__((ext_vector_type(16))) __bf16          v16bf;
typedef __attribute__((ext_vector_type(8)))  float           v8f;
typedef __attribute__((ext_vector_type(4)))  float           f32x4;
typedef __attribute__((ext_vector_type(4)))  unsigned int    u32x4;

union FragBF { v16bf v; u32x4 q[2]; };

__device__ __forceinline__ unsigned int f2bf(float f) {
  unsigned int u = __float_as_uint(f);
  u += 0x7fffu + ((u >> 16) & 1u);   // round to nearest even
  return u >> 16;
}
__device__ __forceinline__ unsigned int pack_bf(float lo, float hi) {
  return f2bf(lo) | (f2bf(hi) << 16);
}

// Branch-free tanh: prefer gfx1250 V_TANH_F32, else v_exp + v_rcp identity.
__device__ __forceinline__ float fast_tanh(float x) {
#if __has_builtin(__builtin_amdgcn_tanhf)
  return __builtin_amdgcn_tanhf(x);
#else
  float xc = fminf(fmaxf(x, -12.0f), 12.0f);
  float e2 = __expf(2.0f * xc);                       // v_exp_f32
  return (e2 - 1.0f) * __builtin_amdgcn_rcpf(e2 + 1.0f);  // v_rcp_f32
#endif
}

// ---------------------------------------------------------------- Wv -> bf16
__global__ __launch_bounds__(256)
void wv_convert_kernel(const float* __restrict__ wv, unsigned int* __restrict__ out) {
  const int gid = blockIdx.x * 256 + threadIdx.x;   // 8 floats -> 4 dwords each
  const f32x4* src = (const f32x4*)(wv + gid * 8);
  f32x4 f0 = src[0], f1 = src[1];
  u32x4 p;
  p[0] = pack_bf(f0.x, f0.y);
  p[1] = pack_bf(f0.z, f0.w);
  p[2] = pack_bf(f1.x, f1.y);
  p[3] = pack_bf(f1.z, f1.w);
  *(u32x4*)(out + gid * 4) = p;
}

// -------------------------------------------------- qproj = query@Wq.T + bias
__global__ __launch_bounds__(256)
void qproj_kernel(const float* __restrict__ query, const float* __restrict__ Wq,
                  const float* __restrict__ bias, float* __restrict__ qproj) {
  const int gw   = (blockIdx.x * 256 + threadIdx.x) >> 5;  // one wave per (b,a)
  const int lane = threadIdx.x & 31;
  const int b = gw >> 10;
  const int a = gw & (Av - 1);
  const f32x4* q4 = (const f32x4*)(query + b * Dv);
  const f32x4* w4 = (const f32x4*)(Wq + (size_t)a * Dv);
  float acc = 0.f;
  #pragma unroll
  for (int i = 0; i < 8; ++i) {
    f32x4 qv = q4[lane + i * 32];
    f32x4 wv = w4[lane + i * 32];
    acc += qv.x * wv.x + qv.y * wv.y + qv.z * wv.z + qv.w * wv.w;
  }
  #pragma unroll
  for (int off = 16; off > 0; off >>= 1) acc += __shfl_xor(acc, off, 32);
  if (lane == 0) qproj[b * Av + a] = acc + bias[a];
}

// -------- main fused kernel: bf16 WMMA GEMM + conv-loc + tanh + fc reduction
__global__ __launch_bounds__(256)
void energy_kernel(const float* __restrict__ value,
                   const unsigned short* __restrict__ wv_bf,
                   const float* __restrict__ qproj,
                   const float* __restrict__ last_attn,
                   const float* __restrict__ conv_w,
                   const float* __restrict__ conv_b,
                   const float* __restrict__ fc_w,
                   const float* __restrict__ fc_b,
                   float* __restrict__ energy) {
  extern __shared__ unsigned char smem[];
  unsigned short* tile = (unsigned short*)smem;            // TM x LDSS bf16
  float* la_s = (float*)(smem + TM * LDSS * 2);            // TM+2 (t-1..t+TM)
  float* e_s  = la_s + (TM + 2);                           // TM energy accum

  const int b    = blockIdx.y;
  const int t0   = blockIdx.x * TM;
  const int tid  = threadIdx.x;
  const int lane = tid & 31;
  const int wave = tid >> 5;

  if (tid < TM) e_s[tid] = 0.0f;
  if (tid < TM + 2) {
    int t = t0 + tid - 1;
    la_s[tid] = (t >= 0 && t < Tv) ? last_attn[b * Tv + t] : 0.0f;
  }

  // stage value tile (fp32 HBM -> bf16 LDS), one pass, 16B chunks
  {
    const float* vb = value + ((size_t)b * Tv + t0) * Dv;
    #pragma unroll
    for (int j = 0; j < 16; ++j) {
      int chunk = j * 256 + tid;         // 4096 chunks of 8 elements
      int m = chunk >> 7;
      int c = chunk & 127;
      const f32x4* src = (const f32x4*)(vb + (size_t)m * Dv + c * 8);
      f32x4 f0 = src[0], f1 = src[1];
      u32x4 p;
      p[0] = pack_bf(f0.x, f0.y);
      p[1] = pack_bf(f0.z, f0.w);
      p[2] = pack_bf(f1.x, f1.y);
      p[3] = pack_bf(f1.z, f1.w);
      *(u32x4*)(tile + m * LDSS + c * 8) = p;
    }
  }
  __syncthreads();

  const int wgrp = wave >> 2;            // row group: rows 0-15 or 16-31
  const int wn   = wave & 3;             // column group: 256-wide strip
  const int lh   = lane >> 4;            // half-wave
  const int l15  = lane & 15;
  // A-frag per ISA layout: lanes 0-15 K{0..7,16..23}, lanes 16-31 K{8..15,24..31}
  const unsigned char* arow = (const unsigned char*)tile
      + (wgrp * 16 + l15) * (LDSS * 2) + lh * 16;

  float eacc[8];
  #pragma unroll
  for (int r = 0; r < 8; ++r) eacc[r] = 0.0f;

  #pragma unroll 1
  for (int noff = 0; noff < 256; noff += 16) {
    const int acol = wn * 256 + noff + l15;
    // B-frag: column = l15, lanes 0-15 K{0..7,16..23}, lanes 16-31 K{8..15,24..31}
    const unsigned short* bcol = wv_bf + (size_t)acol * Dv + lh * 8;

    v8f acc = {0.f, 0.f, 0.f, 0.f, 0.f, 0.f, 0.f, 0.f};

    // two-stage software pipeline: loads for stage i+1 in flight during wmma i
    FragBF A0, B0, A1, B1;
    A0.q[0] = *(const u32x4*)(arow + 0);
    A0.q[1] = *(const u32x4*)(arow + 32);
    B0.q[0] = *(const u32x4*)(bcol + 0);
    B0.q[1] = *(const u32x4*)(bcol + 16);
    #pragma unroll
    for (int k = 0; k < Dv; k += 64) {
      A1.q[0] = *(const u32x4*)(arow + (k + 32) * 2);
      A1.q[1] = *(const u32x4*)(arow + (k + 32) * 2 + 32);
      B1.q[0] = *(const u32x4*)(bcol + (k + 32));
      B1.q[1] = *(const u32x4*)(bcol + (k + 32) + 16);
      acc = __builtin_amdgcn_wmma_f32_16x16x32_bf16(false, A0.v, false, B0.v,
                                                    (short)0, acc, false, false);
      if (k + 64 < Dv) {
        A0.q[0] = *(const u32x4*)(arow + (k + 64) * 2);
        A0.q[1] = *(const u32x4*)(arow + (k + 64) * 2 + 32);
        B0.q[0] = *(const u32x4*)(bcol + (k + 64));
        B0.q[1] = *(const u32x4*)(bcol + (k + 64) + 16);
      }
      acc = __builtin_amdgcn_wmma_f32_16x16x32_bf16(false, A1.v, false, B1.v,
                                                    (short)0, acc, false, false);
    }

    // fused epilogue: qproj(+bias) + conv loc + tanh, scale by fc_w, row-reduce
    const float qp  = qproj[b * Av + acol];
    const float cw0 = conv_w[acol * 3 + 0];
    const float cw1 = conv_w[acol * 3 + 1];
    const float cw2 = conv_w[acol * 3 + 2];
    const float cb  = conv_b[acol];
    const float fw  = fc_w[acol];
    #pragma unroll
    for (int r = 0; r < 8; ++r) {
      const int ml = wgrp * 16 + lh * 8 + r;   // C layout: row r (lanes<16) / r+8
      const float loc = cb + cw0 * la_s[ml] + cw1 * la_s[ml + 1] + cw2 * la_s[ml + 2];
      float e = fast_tanh(acc[r] + qp + loc) * fw;
      e += __shfl_xor(e, 1, 32);
      e += __shfl_xor(e, 2, 32);
      e += __shfl_xor(e, 4, 32);
      e += __shfl_xor(e, 8, 32);
      eacc[r] += e;                            // valid on lanes 0 and 16
    }
  }

  if (l15 == 0) {
    #pragma unroll
    for (int r = 0; r < 8; ++r)
      atomicAdd(&e_s[wgrp * 16 + lh * 8 + r], eacc[r]);
  }
  __syncthreads();
  if (tid < TM) energy[b * Tv + t0 + tid] = e_s[tid] + fc_b[0];
}

// ---------------------------------------------------------------- softmax(T)
__global__ __launch_bounds__(256)
void softmax_kernel(const float* __restrict__ energy, float* __restrict__ attn) {
  __shared__ float red[256];
  const int b = blockIdx.x, tid = threadIdx.x;
  const float* e = energy + b * Tv;
  float v[8];
  float m = -3.402823466e38f;
  #pragma unroll
  for (int i = 0; i < 8; ++i) { v[i] = e[tid + i * 256]; m = fmaxf(m, v[i]); }
  red[tid] = m; __syncthreads();
  for (int s = 128; s > 0; s >>= 1) {
    if (tid < s) red[tid] = fmaxf(red[tid], red[tid + s]);
    __syncthreads();
  }
  m = red[0]; __syncthreads();
  float sum = 0.f;
  #pragma unroll
  for (int i = 0; i < 8; ++i) { v[i] = __expf(v[i] - m); sum += v[i]; }
  red[tid] = sum; __syncthreads();
  for (int s = 128; s > 0; s >>= 1) {
    if (tid < s) red[tid] += red[tid + s];
    __syncthreads();
  }
  const float inv = 1.0f / red[0];
  #pragma unroll
  for (int i = 0; i < 8; ++i) attn[b * Tv + tid + i * 256] = v[i] * inv;
}

// --------------------------------------------------- context = attn @ value
__global__ __launch_bounds__(256)
void context_kernel(const float* __restrict__ attn, const float* __restrict__ value,
                    float* __restrict__ ctx) {
  const int b = blockIdx.y;
  const int d = blockIdx.x * 256 + threadIdx.x;
  const float* vb = value + (size_t)b * Tv * Dv + d;
  const float* ab = attn + b * Tv;
  float acc = 0.f;
  #pragma unroll 8
  for (int t = 0; t < Tv; ++t) acc = fmaf(ab[t], vb[(size_t)t * Dv], acc);
  ctx[b * Dv + d] = acc;
}

extern "C" void kernel_launch(void* const* d_in, const int* in_sizes, int n_in,
                              void* d_out, int out_size, void* d_ws, size_t ws_size,
                              hipStream_t stream) {
  (void)in_sizes; (void)n_in; (void)out_size; (void)ws_size;
  const float* query     = (const float*)d_in[0];
  const float* value     = (const float*)d_in[1];
  const float* last_attn = (const float*)d_in[2];
  const float* conv_w    = (const float*)d_in[3];
  const float* conv_b    = (const float*)d_in[4];
  const float* Wq        = (const float*)d_in[5];
  const float* Wv        = (const float*)d_in[6];
  const float* bias      = (const float*)d_in[7];
  const float* fc_w      = (const float*)d_in[8];
  const float* fc_b      = (const float*)d_in[9];

  float* ctx_out  = (float*)d_out;              // (B,1,D) first
  float* attn_out = (float*)d_out + Bv * Dv;    // then (B,T)

  unsigned char* ws = (unsigned char*)d_ws;
  unsigned int*   wv_bf32 = (unsigned int*)ws;                            // 2 MB bf16 Wv
  unsigned short* wv_bf   = (unsigned short*)ws;
  float* qproj  = (float*)(ws + (size_t)Av * Dv * 2);                     // 128 KB
  float* energy = (float*)(ws + (size_t)Av * Dv * 2 + (size_t)Bv * Av * 4); // 256 KB

  hipFuncSetAttribute((const void*)energy_kernel,
                      hipFuncAttributeMaxDynamicSharedMemorySize, ENERGY_LDS);

  wv_convert_kernel<<<(Av * Dv) / (256 * 8), 256, 0, stream>>>(Wv, wv_bf32);
  qproj_kernel<<<(Bv * Av) / 8, 256, 0, stream>>>(query, Wq, bias, qproj);
  energy_kernel<<<dim3(Tv / TM, Bv), 256, ENERGY_LDS, stream>>>(
      value, wv_bf, qproj, last_attn, conv_w, conv_b, fc_w, fc_b, energy);
  softmax_kernel<<<Bv, 256, 0, stream>>>(energy, attn_out);
  context_kernel<<<dim3(Dv / 256, Bv), 256, 0, stream>>>(attn_out, value, ctx_out);
}
